// HMGRU_TNA_87557203297017
// MI455X (gfx1250) — matrix operand order repaired and checked by reading, not compiled, not run on hardware
//
#include <hip/hip_runtime.h>

// Problem constants (match reference)
#define BB 8
#define TT 8192
#define CC 256
#define HH 4
#define DD 64
#define KS 9
#define PADW 4

// GEMM tiling
#define MT 256            // rows per block (8 waves x 32 rows)
#define NT 64             // weight rows (output cols) staged per iteration
#define PA 264            // LDS row pitch in halves (multiple of 8 -> 16B aligned rows)

// Attention tiling
#define AT 256            // t's per block
#define KP 68             // LDS pitch (halves) for attention tiles

typedef __attribute__((ext_vector_type(16))) _Float16 v16h;
typedef __attribute__((ext_vector_type(8)))  _Float16 v8h;
typedef __attribute__((ext_vector_type(4)))  _Float16 v4h;
typedef __attribute__((ext_vector_type(8)))  float    v8f;
typedef __attribute__((ext_vector_type(4)))  float    f32x4;

// ---------------------------------------------------------------------------
// WMMA fragment loaders (layouts per CDNA5 ISA 7.12.2, wave32)
// A 16x32 f16: lane holds row M=lane%16; halves 0-7 -> K = kb + (lane<16?0:8)+h
//              halves 8-15 -> K = kb + 16 + (lane<16?0:8)+h
// ---------------------------------------------------------------------------
__device__ __forceinline__ v16h frag_a(const _Float16* sA, int rowBase, int kb, int lane) {
    int r  = rowBase + (lane & 15);
    int ko = kb + ((lane & 16) >> 1);            // +8 for upper half-wave
    const _Float16* p = sA + r * PA + ko;
    v8h lo = *(const v8h*)(p);
    v8h hi = *(const v8h*)(p + 16);
    return __builtin_shufflevector(lo, hi, 0,1,2,3,4,5,6,7,8,9,10,11,12,13,14,15);
}

// B 32x16 f16 (K x N): lane holds col N=lane%16; halves 0-7 -> K = kb + (lane<16?0:8)+h, halves 8-15 -> K = kb + 16 + (lane<16?0:8)+h
// sB stored as [n][k] (i.e. W[n,k]), so contiguous along k.
__device__ __forceinline__ v16h frag_b(const _Float16* sB, int nSub, int kb, int lane) {
    int n  = nSub + (lane & 15);
    int ko = kb + ((lane & 16) >> 1);            // +8 for upper half-wave
    const _Float16* p = sB + n * PA + ko;
    v8h lo = *(const v8h*)(p);
    v8h hi = *(const v8h*)(p + 16);
    return __builtin_shufflevector(lo, hi, 0,1,2,3,4,5,6,7,8,9,10,11,12,13,14,15);
}

// ---------------------------------------------------------------------------
// Cooperative tile loaders: global (rows x 256) -> LDS f16 with pitch PA
// ---------------------------------------------------------------------------
__device__ __forceinline__ void load_tile_f32(_Float16* dst, const float* __restrict__ src,
                                              int nrows, int tid) {
    int chunks = nrows * 32;                     // 8-float units per row: 256/8
    for (int u = tid; u < chunks; u += 256) {
        int row = u >> 5;
        int col = (u & 31) << 3;
        const float* s = src + (size_t)row * CC + col;
        f32x4 x0 = *(const f32x4*)(s);
        f32x4 x1 = *(const f32x4*)(s + 4);
        v8h h;
        h[0]=(_Float16)x0[0]; h[1]=(_Float16)x0[1]; h[2]=(_Float16)x0[2]; h[3]=(_Float16)x0[3];
        h[4]=(_Float16)x1[0]; h[5]=(_Float16)x1[1]; h[6]=(_Float16)x1[2]; h[7]=(_Float16)x1[3];
        *(v8h*)(dst + row * PA + col) = h;
    }
}

__device__ __forceinline__ void load_tile_f16(_Float16* dst, const _Float16* __restrict__ src,
                                              int nrows, int tid) {
    int chunks = nrows * 32;
    for (int u = tid; u < chunks; u += 256) {
        int row = u >> 5;
        int col = (u & 31) << 3;
        *(v8h*)(dst + row * PA + col) = *(const v8h*)(src + (size_t)row * CC + col);
    }
}

// ---------------------------------------------------------------------------
// Kernel 1: fused q/k/v projection. x (f32) -> q16/k16/v16 (f16), y = x @ W^T + b
// 8 waves, each owns a 32(M) x 64(N) tile: 2 A-frags + 4 B-frags per K-chunk
// feed 8 WMMAs (1.5 ds_load_b128 per WMMA).
// ---------------------------------------------------------------------------
__global__ __launch_bounds__(256) void qkv_kernel(
    const float* __restrict__ x,
    const float* __restrict__ Wq, const float* __restrict__ bq,
    const float* __restrict__ Wk, const float* __restrict__ bk,
    const float* __restrict__ Wv, const float* __restrict__ bv,
    _Float16* __restrict__ q16, _Float16* __restrict__ k16, _Float16* __restrict__ v16)
{
    __shared__ __align__(16) _Float16 sA[MT * PA];   // 256x256 x-tile (f16), 132KB
    __shared__ __align__(16) _Float16 sB[NT * PA];   // 64x256 weight slab (f16), 33KB

    const int tid  = threadIdx.x;
    const int lane = tid & 31;
    const int wv   = tid >> 5;                       // 8 waves
    const size_t gm0 = (size_t)blockIdx.x * MT;

    load_tile_f32(sA, x + gm0 * CC, MT, tid);

    const float*  Ws[3] = { Wq, Wk, Wv };
    const float*  bs[3] = { bq, bk, bv };
    _Float16*     Os[3] = { q16, k16, v16 };

    const int rowBase = wv * 32;
    const int nCol = lane & 15;
    const int rOff = (lane & 16) >> 1;               // 0 or 8 (C/D layout)

    for (int p = 0; p < 3; ++p) {
        const float* W = Ws[p];
        const float* bias = bs[p];
        _Float16* O = Os[p];
        for (int nt = 0; nt < 4; ++nt) {
            const int n0 = nt * 64;
            __syncthreads();
            load_tile_f32(sB, W + (size_t)n0 * CC, NT, tid);
            // prefetch next weight slab into L2/WGP$ while we compute this one
            const float* nextW = (nt < 3) ? (W + (size_t)(n0 + 64) * CC)
                               : (p < 2) ? Ws[p + 1] : (const float*)0;
            if (nextW) __builtin_prefetch(nextW + (size_t)tid * 64, 0, 1);
            __syncthreads();

            v8f acc[2][4] = {};
            #pragma unroll
            for (int kc = 0; kc < 8; ++kc) {
                const int kb = kc * 32;
                v16h a0 = frag_a(sA, rowBase,      kb, lane);
                v16h a1 = frag_a(sA, rowBase + 16, kb, lane);
                #pragma unroll
                for (int ns = 0; ns < 4; ++ns) {
                    v16h bf = frag_b(sB, ns * 16, kb, lane);
                    acc[0][ns] = __builtin_amdgcn_wmma_f32_16x16x32_f16(false, a0, false, bf, (short)0, acc[0][ns], false, false);
                    acc[1][ns] = __builtin_amdgcn_wmma_f32_16x16x32_f16(false, a1, false, bf, (short)0, acc[1][ns], false, false);
                }
            }

            #pragma unroll
            for (int ns = 0; ns < 4; ++ns) {
                const int   cg = n0 + ns * 16 + nCol;
                const float bn = bias[cg];
                #pragma unroll
                for (int mi = 0; mi < 2; ++mi) {
                    #pragma unroll
                    for (int r = 0; r < 8; ++r) {
                        size_t row = gm0 + rowBase + mi * 16 + rOff + r;
                        O[row * CC + cg] = (_Float16)(acc[mi][ns][r] + bn);
                    }
                }
            }
        }
    }
}

// ---------------------------------------------------------------------------
// Kernel 2: sliding-window attention (KS=9, replicate padding), f16 in/out
// One thread per t; block covers AT t's for one (b, h).
// ---------------------------------------------------------------------------
__global__ __launch_bounds__(256) void attn_kernel(
    const _Float16* __restrict__ q16, const _Float16* __restrict__ k16,
    const _Float16* __restrict__ v16, _Float16* __restrict__ y16)
{
    __shared__ __align__(16) _Float16 sK[(AT + 2 * PADW) * KP];
    __shared__ __align__(16) _Float16 sV[(AT + 2 * PADW) * KP];
    __shared__ __align__(16) _Float16 sQ[AT * KP];

    const int tid = threadIdx.x;
    const int nBlkT = TT / AT;                       // 32
    const int tb = blockIdx.x % nBlkT;
    const int h  = (blockIdx.x / nBlkT) % HH;
    const int b  =  blockIdx.x / (nBlkT * HH);
    const int t0 = tb * AT;
    const size_t baseRow = (size_t)b * TT;
    const size_t colOff  = (size_t)h * DD;

    // stage k/v windows with edge-replication clamp
    for (int u = tid; u < (AT + 2 * PADW) * 16; u += 256) {
        int row  = u >> 4;
        int part = (u & 15) << 2;                    // 4-half (8B) units
        int t = t0 - PADW + row;
        t = t < 0 ? 0 : (t >= TT ? TT - 1 : t);
        size_t g = (baseRow + (size_t)t) * CC + colOff + part;
        *(v4h*)(sK + row * KP + part) = *(const v4h*)(k16 + g);
        *(v4h*)(sV + row * KP + part) = *(const v4h*)(v16 + g);
    }
    for (int u = tid; u < AT * 16; u += 256) {
        int row  = u >> 4;
        int part = (u & 15) << 2;
        size_t g = (baseRow + (size_t)(t0 + row)) * CC + colOff + part;
        *(v4h*)(sQ + row * KP + part) = *(const v4h*)(q16 + g);
    }
    __syncthreads();

    const int tl = tid;

    // hoist this thread's q row into registers (reused 9x)
    float qv[DD];
    {
        const _Float16* qrow = sQ + tl * KP;
        #pragma unroll
        for (int d = 0; d < DD; ++d) qv[d] = (float)qrow[d];
    }

    float s[KS];
    #pragma unroll
    for (int j = 0; j < KS; ++j) {
        const _Float16* krow = sK + (tl + j) * KP;
        float acc = 0.f;
        #pragma unroll
        for (int d = 0; d < DD; ++d)
            acc = __builtin_fmaf(qv[d], (float)krow[d], acc);
        s[j] = acc * 0.125f;                         // 1/sqrt(64)
    }

    float m = s[0];
    #pragma unroll
    for (int j = 1; j < KS; ++j) m = fmaxf(m, s[j]);
    float e[KS]; float sum = 0.f;
    #pragma unroll
    for (int j = 0; j < KS; ++j) { e[j] = __expf(s[j] - m); sum += e[j]; }
    const float inv = 1.f / sum;
    #pragma unroll
    for (int j = 0; j < KS; ++j) e[j] *= inv;

    _Float16* yr = y16 + (baseRow + (size_t)(t0 + tl)) * CC + colOff;
    #pragma unroll 8
    for (int d = 0; d < DD; ++d) {
        float acc = 0.f;
        #pragma unroll
        for (int j = 0; j < KS; ++j)
            acc = __builtin_fmaf(e[j], (float)sV[(tl + j) * KP + d], acc);
        yr[d] = (_Float16)acc;
    }
}

// ---------------------------------------------------------------------------
// Kernel 3: output projection. y16 (f16) @ Wo^T + bo -> out (f32)
// ---------------------------------------------------------------------------
__global__ __launch_bounds__(256) void oproj_kernel(
    const _Float16* __restrict__ y16, const float* __restrict__ Wo,
    const float* __restrict__ bo, float* __restrict__ out)
{
    __shared__ __align__(16) _Float16 sA[MT * PA];
    __shared__ __align__(16) _Float16 sB[NT * PA];

    const int tid  = threadIdx.x;
    const int lane = tid & 31;
    const int wv   = tid >> 5;
    const size_t gm0 = (size_t)blockIdx.x * MT;

    load_tile_f16(sA, y16 + gm0 * CC, MT, tid);

    const int rowBase = wv * 32;
    const int nCol = lane & 15;
    const int rOff = (lane & 16) >> 1;

    for (int nt = 0; nt < 4; ++nt) {
        const int n0 = nt * 64;
        __syncthreads();
        load_tile_f32(sB, Wo + (size_t)n0 * CC, NT, tid);
        if (nt < 3) __builtin_prefetch(Wo + (size_t)(n0 + 64) * CC + (size_t)tid * 64, 0, 1);
        __syncthreads();

        v8f acc[2][4] = {};
        #pragma unroll
        for (int kc = 0; kc < 8; ++kc) {
            const int kb = kc * 32;
            v16h a0 = frag_a(sA, rowBase,      kb, lane);
            v16h a1 = frag_a(sA, rowBase + 16, kb, lane);
            #pragma unroll
            for (int ns = 0; ns < 4; ++ns) {
                v16h bf = frag_b(sB, ns * 16, kb, lane);
                acc[0][ns] = __builtin_amdgcn_wmma_f32_16x16x32_f16(false, a0, false, bf, (short)0, acc[0][ns], false, false);
                acc[1][ns] = __builtin_amdgcn_wmma_f32_16x16x32_f16(false, a1, false, bf, (short)0, acc[1][ns], false, false);
            }
        }

        #pragma unroll
        for (int ns = 0; ns < 4; ++ns) {
            const int   cg = n0 + ns * 16 + nCol;
            const float bn = bo[cg];
            #pragma unroll
            for (int mi = 0; mi < 2; ++mi) {
                #pragma unroll
                for (int r = 0; r < 8; ++r) {
                    size_t row = gm0 + rowBase + mi * 16 + rOff + r;
                    out[row * CC + cg] = acc[mi][ns][r] + bn;
                }
            }
        }
    }
}

// ---------------------------------------------------------------------------
extern "C" void kernel_launch(void* const* d_in, const int* in_sizes, int n_in,
                              void* d_out, int out_size, void* d_ws, size_t ws_size,
                              hipStream_t stream) {
    const float* x  = (const float*)d_in[0];
    const float* Wq = (const float*)d_in[1];
    const float* bq = (const float*)d_in[2];
    const float* Wk = (const float*)d_in[3];
    const float* bk = (const float*)d_in[4];
    const float* Wv = (const float*)d_in[5];
    const float* bv = (const float*)d_in[6];
    const float* Wo = (const float*)d_in[7];
    const float* bo = (const float*)d_in[8];
    float* out = (float*)d_out;

    const size_t n = (size_t)BB * TT * CC;           // 16,777,216 elements
    _Float16* q16 = (_Float16*)d_ws;                 // 32 MiB each
    _Float16* k16 = q16 + n;
    _Float16* v16 = k16 + n;
    _Float16* y16 = v16 + n;

    const int gemmBlocks = (BB * TT) / MT;           // 256
    qkv_kernel<<<gemmBlocks, 256, 0, stream>>>(x, Wq, bq, Wk, bk, Wv, bv, q16, k16, v16);

    const int attnBlocks = BB * HH * (TT / AT);      // 1024
    attn_kernel<<<attnBlocks, 256, 0, stream>>>(q16, k16, v16, y16);

    oproj_kernel<<<gemmBlocks, 256, 0, stream>>>(y16, Wo, bo, out);
}
